// AttenWeight_75402445848857
// MI455X (gfx1250) — compile-verified, matrix operand order checked by reading, not run on hardware
//
#include <hip/hip_runtime.h>
#include <hip/hip_bf16.h>

typedef _Float16 half_t;
typedef __attribute__((ext_vector_type(16))) _Float16 v16h;
typedef __attribute__((ext_vector_type(8)))  _Float16 v8h;
typedef __attribute__((ext_vector_type(8)))  float    v8f;

#define NH   8
#define NN   384
#define NL   64
#define DIN  128
#define DOUT 32
#define NROWS (NH*NN*NL)              // 196608
#define KQ_ELEMS ((size_t)NROWS*DOUT) // 6291456 per matrix

__device__ __forceinline__ v8f wmma_f16(v16h a, v16h b, v8f c) {
  return __builtin_amdgcn_wmma_f32_16x16x32_f16(false, a, false, b, (short)0, c, false, false);
}

// Build a 16x32-f16 WMMA A/B fragment from a row-major f16 row (LDS or global).
// Per ISA 7.12.2: lane-half hf=0 needs K{0..7,16..23}, hf=1 needs K{8..15,24..31}
// -> two contiguous 16B chunks at +hf*8 and +hf*8+16 halves.
__device__ __forceinline__ v16h frag_from_row(const half_t* rowp, int hf) {
  v8h lo = *(const v8h*)(rowp + hf*8);
  v8h hi = *(const v8h*)(rowp + hf*8 + 16);
  v16h r;
#pragma unroll
  for (int e = 0; e < 8; ++e) { r[e] = lo[e]; r[e+8] = hi[e]; }
  return r;
}

// ---------------------------------------------------------------------------
// Kernel 0: one-shot f32 -> f16 conversion of the three weight matrices into
// workspace so the streaming kernel never does VALU converts for weights.
// ---------------------------------------------------------------------------
__global__ __launch_bounds__(256)
void prep_weights_kernel(const float* __restrict__ proj_w,
                         const float* __restrict__ wk,
                         const float* __restrict__ wq,
                         half_t* __restrict__ wout)
{
  int t = blockIdx.x * 256 + threadIdx.x;
  if (t < DOUT*DIN)                 wout[t] = (half_t)proj_w[t];
  else if (t < DOUT*DIN + DOUT*DOUT)   wout[t] = (half_t)wk[t - DOUT*DIN];
  else if (t < DOUT*DIN + 2*DOUT*DOUT) wout[t] = (half_t)wq[t - DOUT*DIN - DOUT*DOUT];
}

// ---------------------------------------------------------------------------
// Kernel 1: fused LayerNorm + proj(128->32, +bias) + k/q proj (32->32),
// one streaming pass over x.  4 waves/block, one 16-row tile per wave.
// ---------------------------------------------------------------------------
__global__ __launch_bounds__(128)
void ln_proj_kq_kernel(const float* __restrict__ x,
                       const float* __restrict__ ln_w,  const float* __restrict__ ln_b,
                       const half_t* __restrict__ pwh,  const float* __restrict__ proj_b,
                       const half_t* __restrict__ wkh,  const half_t* __restrict__ wqh,
                       half_t* __restrict__ kws, half_t* __restrict__ qws)
{
  __shared__ __align__(16) float  xs[4][16*DIN];   // 32 KB
  __shared__ __align__(16) half_t xn[4][16*DIN];   // 16 KB
  __shared__ __align__(16) half_t xp[4][16*DOUT];  //  4 KB
  __shared__ float s_lnw[DIN], s_lnb[DIN];

  const int tid  = threadIdx.x;
  const int w    = tid >> 5;
  const int lane = tid & 31;
  const int n0   = lane & 15;      // N / M index within tile
  const int hf   = lane >> 4;      // lane half

  if (tid < DIN) { s_lnw[tid] = ln_w[tid]; s_lnb[tid] = ln_b[tid]; }

  const int tile    = blockIdx.x * 4 + w;
  const int rowBase = tile * 16;

  // --- load 16x128 f32 tile (coalesced b128) ---
  const float4* xsrc = (const float4*)(x + (size_t)rowBase * DIN);
  float4*       xdst = (float4*)xs[w];
#pragma unroll
  for (int it = 0; it < 16; ++it) xdst[lane + it*32] = xsrc[lane + it*32];
  __syncthreads();

  // --- LayerNorm stats: 2 lanes per row ---
  const float* xr = &xs[w][n0*DIN + hf*64];
  float s = 0.f, s2 = 0.f;
#pragma unroll
  for (int t = 0; t < 64; ++t) { float v = xr[t]; s += v; s2 += v*v; }
  s  += __shfl_xor(s, 16, 32);
  s2 += __shfl_xor(s2, 16, 32);
  const float mu   = s * (1.f/128.f);
  const float var  = s2 * (1.f/128.f) - mu*mu;
  const float rstd = rsqrtf(var + 1e-5f);

  // --- normalize + affine -> f16 ---
#pragma unroll
  for (int t = 0; t < 64; ++t) {
    int c = hf*64 + t;
    float v = (xs[w][n0*DIN + c] - mu) * rstd * s_lnw[c] + s_lnb[c];
    xn[w][n0*DIN + c] = (half_t)v;
  }
  __syncthreads();

  // --- proj: xp = xn @ proj_w^T + proj_b  (16x128 @ 128x32) ---
  v16h bp[2][4];
#pragma unroll
  for (int nt = 0; nt < 2; ++nt)
#pragma unroll
    for (int kb = 0; kb < 4; ++kb)
      bp[nt][kb] = frag_from_row(pwh + (n0 + nt*16)*DIN + kb*32, hf);

  v8f c0, c1;
  { float b0 = proj_b[n0], b1 = proj_b[n0 + 16];
#pragma unroll
    for (int r = 0; r < 8; ++r) { c0[r] = b0; c1[r] = b1; } }

#pragma unroll
  for (int kb = 0; kb < 4; ++kb) {
    v16h a = frag_from_row(&xn[w][n0*DIN + kb*32], hf);
    c0 = wmma_f16(a, bp[0][kb], c0);
    c1 = wmma_f16(a, bp[1][kb], c1);
  }

  // stash xp tile (row-major f16)
#pragma unroll
  for (int r = 0; r < 8; ++r) {
    int m = r + hf*8;
    xp[w][m*DOUT + n0]      = (half_t)c0[r];
    xp[w][m*DOUT + n0 + 16] = (half_t)c1[r];
  }
  __syncthreads();

  // --- k = xp @ wk^T , q = xp @ wq^T  (16x32 @ 32x32) ---
  v16h bk0 = frag_from_row(wkh + n0*DOUT, hf);
  v16h bk1 = frag_from_row(wkh + (n0+16)*DOUT, hf);
  v16h bq0 = frag_from_row(wqh + n0*DOUT, hf);
  v16h bq1 = frag_from_row(wqh + (n0+16)*DOUT, hf);

  v16h a2 = frag_from_row(&xp[w][n0*DOUT], hf);
  v8f z = {};
  v8f ck0 = wmma_f16(a2, bk0, z);
  v8f ck1 = wmma_f16(a2, bk1, z);
  v8f cq0 = wmma_f16(a2, bq0, z);
  v8f cq1 = wmma_f16(a2, bq1, z);

  // scatter-store K/Q as f16 in [h][l][i][d] so kernel 2 loads coalesced
#pragma unroll
  for (int r = 0; r < 8; ++r) {
    int gm = rowBase + r + hf*8;
    int h  = gm / (NN*NL);
    int i  = (gm >> 6) % NN;
    int l  = gm & 63;
    size_t base = (((size_t)(h*NL + l))*NN + i) * DOUT;
    kws[base + n0]      = (half_t)ck0[r];
    kws[base + n0 + 16] = (half_t)ck1[r];
    qws[base + n0]      = (half_t)cq0[r];
    qws[base + n0 + 16] = (half_t)cq1[r];
  }
}

// ---------------------------------------------------------------------------
// Kernel 2: per (h,l) pair: scores = K @ Q^T * scale (+eps), softmax over i
// (online, per column j), mean over j.  One block per (h,l), all in LDS.
// K/Q tiles are brought in with ASYNC global->LDS copies (ASYNCcnt path).
// ---------------------------------------------------------------------------
#define SCALING 0.17677669529663687f  // 1/sqrt(32)
#define SM_EPS  1e-6f

__global__ __launch_bounds__(256)
void atten_kernel(const half_t* __restrict__ kws, const half_t* __restrict__ qws,
                  float* __restrict__ out)
{
  __shared__ __align__(16) half_t Kt[NN*DOUT];  // 24 KB
  __shared__ __align__(16) half_t Qt[NN*DOUT];  // 24 KB
  __shared__ float outAcc[NN];

  const int tid = threadIdx.x;
  const size_t base = (size_t)blockIdx.x * (NN*DOUT);

  // ---- async global -> LDS tile fetch (GLOBAL_LOAD_ASYNC_TO_LDS_B128) ----
  {
    const void* ksrc = (const void*)(kws + base);
    const void* qsrc = (const void*)(qws + base);
    unsigned lk = (unsigned)(size_t)(void*)Kt;   // addrspace(3) offset in low bits
    unsigned lq = (unsigned)(size_t)(void*)Qt;
#pragma unroll
    for (int it = 0; it < 6; ++it) {             // 256 lanes * 6 * 16B = 24 KB
      unsigned off = (unsigned)(tid + it*256) * 16u;
      asm volatile("global_load_async_to_lds_b128 %0, %1, %2"
                   :: "v"(lk + off), "v"(off), "s"(ksrc) : "memory");
      asm volatile("global_load_async_to_lds_b128 %0, %1, %2"
                   :: "v"(lq + off), "v"(off), "s"(qsrc) : "memory");
    }
    asm volatile("s_wait_asynccnt 0x0" ::: "memory");
  }
  outAcc[tid] = 0.f;
  if (tid < NN - 256) outAcc[256 + tid] = 0.f;
  __syncthreads();

  const int w    = tid >> 5;
  const int lane = tid & 31;
  const int n0   = lane & 15;
  const int hf   = lane >> 4;
  v8f z = {};

  for (int js = w; js < 24; js += 8) {       // each wave owns 3 j-strips
    const int j0 = js * 16;
    v16h b = frag_from_row(&Qt[(j0 + n0)*DOUT], hf);  // B[d][j] = Q[j][d]

    // ---- pass 1: online column softmax stats (max m, denom d per column) ----
    float mrun = -__builtin_inff();
    float drun = 0.f;
    for (int ti = 0; ti < 24; ++ti) {
      v16h a = frag_from_row(&Kt[(ti*16 + n0)*DOUT], hf);
      v8f  c = wmma_f16(a, b, z);
      float sc[8], tmax = -__builtin_inff();
#pragma unroll
      for (int r = 0; r < 8; ++r) { sc[r] = c[r]*SCALING + SM_EPS; tmax = fmaxf(tmax, sc[r]); }
      tmax = fmaxf(tmax, __shfl_xor(tmax, 16, 32));   // combine M halves
      float mnew = fmaxf(mrun, tmax);
      float psum = 0.f;
#pragma unroll
      for (int r = 0; r < 8; ++r) psum += __expf(sc[r] - mnew);
      psum += __shfl_xor(psum, 16, 32);
      drun = drun * __expf(mrun - mnew) + psum;
      mrun = mnew;
    }
    const float inv_d = 1.f / drun;

    // ---- pass 2: probabilities, reduce over j (N) into per-row accumulator ----
    for (int ti = 0; ti < 24; ++ti) {
      v16h a = frag_from_row(&Kt[(ti*16 + n0)*DOUT], hf);
      v8f  c = wmma_f16(a, b, z);
      float p[8];
#pragma unroll
      for (int r = 0; r < 8; ++r)
        p[r] = __expf(c[r]*SCALING + SM_EPS - mrun) * inv_d;
#pragma unroll
      for (int mk = 1; mk < 16; mk <<= 1)
#pragma unroll
        for (int r = 0; r < 8; ++r) p[r] += __shfl_xor(p[r], mk, 32);
      if (n0 == 0) {
#pragma unroll
        for (int r = 0; r < 8; ++r)
          atomicAdd(&outAcc[ti*16 + hf*8 + r], p[r]);
      }
    }
  }
  __syncthreads();

  const int h = blockIdx.x >> 6;
  const int l = blockIdx.x & 63;
  for (int t = tid; t < NN; t += 256)
    out[((size_t)(h*NN + t))*NL + l] = outAcc[t] * (1.f / 384.f);
}

// ---------------------------------------------------------------------------
extern "C" void kernel_launch(void* const* d_in, const int* in_sizes, int n_in,
                              void* d_out, int out_size, void* d_ws, size_t ws_size,
                              hipStream_t stream) {
  const float* x      = (const float*)d_in[0];
  const float* ln_w   = (const float*)d_in[1];
  const float* ln_b   = (const float*)d_in[2];
  const float* proj_w = (const float*)d_in[3];
  const float* proj_b = (const float*)d_in[4];
  const float* wk     = (const float*)d_in[5];
  const float* wq     = (const float*)d_in[6];

  half_t* kws = (half_t*)d_ws;
  half_t* qws = kws + KQ_ELEMS;           // 2 * 12 MB of f16 scratch
  half_t* wgt = qws + KQ_ELEMS;           // f16 weights: proj_w | wk | wq
  half_t* pwh = wgt;
  half_t* wkh = pwh + DOUT*DIN;
  half_t* wqh = wkh + DOUT*DOUT;

  prep_weights_kernel<<<(DOUT*DIN + 2*DOUT*DOUT + 255)/256, 256, 0, stream>>>(
      proj_w, wk, wq, wgt);
  // 196608 rows / 16 per wave / 4 waves per block
  ln_proj_kq_kernel<<<NROWS/64, 128, 0, stream>>>(x, ln_w, ln_b, pwh, proj_b,
                                                  wkh, wqh, kws, qws);
  // one block per (h,l) pair
  atten_kernel<<<NH*NL, 256, 0, stream>>>(kws, qws, (float*)d_out);
}
